// EncoderBlock_26714696581260
// MI455X (gfx1250) — matrix operand order repaired and checked by reading, not compiled, not run on hardware
//
#include <hip/hip_runtime.h>

// ---------------- problem constants ----------------
constexpr int NN   = 50000;          // nodes
constexpr int EE   = 800000;         // edges
constexpr int EN   = EE + NN;        // edges + self loops
constexpr int INF_ = 256;            // in channels
constexpr int OUTF = 256;            // out channels
constexpr int HH   = 4;              // heads
constexpr int CC   = OUTF / HH;      // 64 per-head dim
constexpr int KK   = 25000;          // kept nodes
constexpr float NEG_SLOPE = 0.2f;

// ---------------- WMMA types ----------------
typedef __attribute__((ext_vector_type(16))) __bf16 v16bf;
typedef __attribute__((ext_vector_type(8)))  float  v8f;

__device__ __forceinline__ unsigned f2bf(float f) {
  unsigned u = __float_as_uint(f);
  u += 0x7FFFu + ((u >> 16) & 1u);      // round-to-nearest-even
  return u >> 16;
}

// monotone float<->int order mapping (for atomicMax on signed int)
__device__ __forceinline__ int f32_order(float f) {
  int i = __float_as_int(f);
  return i ^ ((i >> 31) & 0x7FFFFFFF);
}
__device__ __forceinline__ float order_f32(int i) {
  return __int_as_float(i ^ ((i >> 31) & 0x7FFFFFFF));
}

// =====================================================================
// Kernel 0: f32 -> bf16 pre-conversion (8 elems / thread, packed stores)
// =====================================================================
__global__ void conv_bf16(const float* __restrict__ src,
                          unsigned short* __restrict__ dst, int n8) {
  int t = blockIdx.x * blockDim.x + threadIdx.x;
  if (t >= n8) return;
  const float4 f0 = ((const float4*)src)[(size_t)t * 2];
  const float4 f1 = ((const float4*)src)[(size_t)t * 2 + 1];
  uint4 o;
  o.x = f2bf(f0.x) | (f2bf(f0.y) << 16);
  o.y = f2bf(f0.z) | (f2bf(f0.w) << 16);
  o.z = f2bf(f1.x) | (f2bf(f1.y) << 16);
  o.w = f2bf(f1.z) | (f2bf(f1.w) << 16);
  ((uint4*)dst)[t] = o;
}

// =====================================================================
// Kernel 1: xh = x @ lin_w^T  via v_wmma_f32_16x16x32_bf16 (bf16 inputs)
// grid = (NN/16, 4), block = 128 (4 waves); wave w -> col tile 4*by+w
// bf16 A tile (16 x 256 = 8KB) staged into LDS with async-to-LDS DMA;
// hot loop is pure ds_load_b128 / global_load_b128 + v_wmma.
// =====================================================================
__global__ void __launch_bounds__(128)
gat_lin_wmma(const unsigned short* __restrict__ xb,
             const unsigned short* __restrict__ wb,
             float* __restrict__ xh) {
  __shared__ unsigned short ldsA[16 * INF_];   // 8 KB bf16
  const int rt   = blockIdx.x;                 // row tile (16 rows)
  const int tid  = threadIdx.x;
  const int wave = tid >> 5;
  const int lane = tid & 31;

  // ---- async stage of bf16 A tile: 512 chunks of 16B, 128 lanes x 4 ----
  {
    const unsigned short* gbase = xb + (size_t)rt * 16 * INF_;
    unsigned ldsBase = (unsigned)(size_t)(&ldsA[0]);   // low 32 bits = LDS offset
#pragma unroll
    for (int it = 0; it < 4; ++it) {
      int chunk = it * 128 + tid;                      // 16B granules
      unsigned laddr = ldsBase + (unsigned)(chunk * 16);
      const unsigned short* g = gbase + chunk * 8;
      asm volatile("global_load_async_to_lds_b128 %0, %1, off"
                   :: "v"(laddr), "v"(g) : "memory");
    }
    asm volatile("s_wait_asynccnt 0" ::: "memory");
  }
  __syncthreads();

  const int ct   = blockIdx.y * 4 + wave;      // col tile 0..15
  const int half = lane >> 4;                  // 0: lanes 0-15, 1: lanes 16-31
  const int mrow = lane & 15;
  const unsigned short* wrow = wb + (size_t)(ct * 16 + mrow) * INF_;
  __builtin_prefetch(wrow, 0, 3);

  union Frag { v16bf v; uint4 q[2]; };
  Frag a, b;
  v8f acc = (v8f)(0.0f);

#pragma unroll
  for (int kk = 0; kk < INF_; kk += 32) {
    // A frag (ISA 7.12.2): lane<16: K=kk..kk+7 & kk+16..kk+23
    //                      lane>=16: K=kk+8..kk+15 & kk+24..kk+31
    const unsigned short* arow = &ldsA[mrow * INF_];
    a.q[0] = *(const uint4*)(arow + kk + half * 8);
    a.q[1] = *(const uint4*)(arow + kk + 16 + half * 8);
    // B frag: lane n<16 holds K=kk..kk+7 & kk+16..kk+23 of col n; lane n+16 holds kk+8..kk+15 & kk+24..kk+31
    const unsigned short* bp = wrow + kk + half * 8;
    b.q[0] = *(const uint4*)(bp);
    b.q[1] = *(const uint4*)(bp + 16);

    acc = __builtin_amdgcn_wmma_f32_16x16x32_bf16(
        false, a.v, false, b.v, (short)0, acc, false, false);
  }

  // C layout: lane<16: VGPR r = (M=r, N=lane); lane>=16: (M=8+r, N=lane-16)
  const int ncol  = ct * 16 + mrow;
  const int mbase = rt * 16 + half * 8;
  float* op = xh + (size_t)mbase * OUTF + ncol;
#pragma unroll
  for (int r = 0; r < 8; ++r) op[(size_t)r * OUTF] = acc[r];
}

// =====================================================================
// Kernel 2: per-node attention logits a_src/a_dst  [N,H]
// =====================================================================
__global__ void att_scores(const float* __restrict__ xh,
                           const float* __restrict__ att_src,
                           const float* __restrict__ att_dst,
                           float* __restrict__ a_src, float* __restrict__ a_dst) {
  int t = blockIdx.x * blockDim.x + threadIdx.x;
  if (t >= NN * HH) return;
  int n = t / HH, h = t % HH;
  const float* v  = xh + (size_t)n * OUTF + h * CC;
  const float* as = att_src + h * CC;
  const float* ad = att_dst + h * CC;
  float s = 0.f, d = 0.f;
#pragma unroll 4
  for (int c = 0; c < CC; ++c) { float xv = v[c]; s += xv * as[c]; d += xv * ad[c]; }
  a_src[t] = s; a_dst[t] = d;
}

// =====================================================================
// Kernel 3: workspace init (m = order(-inf), denom = 0, out_acc = 0)
// =====================================================================
__global__ void init_ws(int* __restrict__ m_i, float* __restrict__ denom,
                        float* __restrict__ out_acc) {
  int t = blockIdx.x * blockDim.x + threadIdx.x;
  if (t < NN * OUTF) out_acc[t] = 0.f;
  if (t < NN * HH) { m_i[t] = f32_order(-__builtin_huge_valf()); denom[t] = 0.f; }
}

// =====================================================================
// Kernel 4: e = leaky_relu(a_src[src]+a_dst[dst]); segment max into m_i
// =====================================================================
__global__ void edge_max(const long long* __restrict__ ei,
                         const float* __restrict__ a_src,
                         const float* __restrict__ a_dst,
                         float* __restrict__ e_buf, int* __restrict__ m_i) {
  int t = blockIdx.x * blockDim.x + threadIdx.x;
  if (t >= EN * HH) return;
  int e = t / HH, h = t % HH;
  int s, d;
  if (e < EE) { s = (int)ei[e]; d = (int)ei[EE + e]; } else { s = d = e - EE; }
  float v = a_src[s * HH + h] + a_dst[d * HH + h];
  v = (v > 0.f) ? v : NEG_SLOPE * v;
  e_buf[t] = v;
  atomicMax(&m_i[d * HH + h], f32_order(v));
}

// =====================================================================
// Kernel 5: ex = exp(e - m[dst]); segment sum into denom
// =====================================================================
__global__ void edge_exp(const long long* __restrict__ ei,
                         const int* __restrict__ m_i,
                         float* __restrict__ e_buf, float* __restrict__ denom) {
  int t = blockIdx.x * blockDim.x + threadIdx.x;
  if (t >= EN * HH) return;
  int e = t / HH, h = t % HH;
  int d = (e < EE) ? (int)ei[EE + e] : (e - EE);
  float m = order_f32(m_i[d * HH + h]);
  float ex = expf(e_buf[t] - m);
  e_buf[t] = ex;
  atomicAdd(&denom[d * HH + h], ex);
}

// =====================================================================
// Kernel 6: out[dst] += alpha * xh[src]  (thread per edge x 4 channels)
// =====================================================================
__global__ void edge_msg(const long long* __restrict__ ei,
                         const float* __restrict__ xh,
                         const float* __restrict__ e_buf,
                         const float* __restrict__ denom,
                         float* __restrict__ out_acc) {
  int t = blockIdx.x * blockDim.x + threadIdx.x;
  if (t >= EN * 64) return;
  int e = t >> 6, g = t & 63;
  int s, d;
  if (e < EE) { s = (int)ei[e]; d = (int)ei[EE + e]; } else { s = d = e - EE; }
  int ch = g * 4;
  int h = ch / CC;
  float alpha = e_buf[e * HH + h] / denom[d * HH + h];
  const float4 xv = *(const float4*)(xh + (size_t)s * OUTF + ch);
  float* op = out_acc + (size_t)d * OUTF + ch;
  atomicAdd(op + 0, xv.x * alpha);
  atomicAdd(op + 1, xv.y * alpha);
  atomicAdd(op + 2, xv.z * alpha);
  atomicAdd(op + 3, xv.w * alpha);
}

// =====================================================================
// Kernel 7: ||pool_w||
// =====================================================================
__global__ void pool_norm(const float* __restrict__ pool_w, float* __restrict__ normp) {
  __shared__ float red[256];
  int c = threadIdx.x;
  float v = pool_w[c];
  red[c] = v * v;
  __syncthreads();
  for (int st = 128; st > 0; st >>= 1) { if (c < st) red[c] += red[c + st]; __syncthreads(); }
  if (c == 0) normp[0] = sqrtf(red[0]);
}

// =====================================================================
// Kernel 8: bias + ELU (in place), score = tanh(out . pool_w / norm)
// one block per node, 256 threads
// =====================================================================
__global__ void finish_node(const float* __restrict__ bias,
                            const float* __restrict__ pool_w,
                            const float* __restrict__ normp,
                            float* __restrict__ out_acc, float* __restrict__ score) {
  __shared__ float red[256];
  int n = blockIdx.x, c = threadIdx.x;
  float v = out_acc[(size_t)n * OUTF + c] + bias[c];
  v = (v > 0.f) ? v : (expf(v) - 1.f);           // ELU
  out_acc[(size_t)n * OUTF + c] = v;
  red[c] = v * pool_w[c];
  __syncthreads();
  for (int st = 128; st > 0; st >>= 1) { if (c < st) red[c] += red[c + st]; __syncthreads(); }
  if (c == 0) score[n] = tanhf(red[0] / normp[0]);
}

// =====================================================================
// Kernel 9: stable descending rank of every score (tiled O(N^2));
// fills new_id, perm, score_k
// =====================================================================
__global__ void topk_rank(const float* __restrict__ score,
                          int* __restrict__ new_id, int* __restrict__ perm,
                          float* __restrict__ score_k) {
  __shared__ unsigned skeys[256];
  int i = blockIdx.x * 256 + threadIdx.x;
  float myscore = 0.f;
  unsigned my = 0u;
  if (i < NN) {
    myscore = score[i];
    my = (unsigned)f32_order(myscore) ^ 0x80000000u;
  }
  int rank = 0;
  for (int base = 0; base < NN; base += 256) {
    int j = base + threadIdx.x;
    skeys[threadIdx.x] = (j < NN) ? ((unsigned)f32_order(score[j]) ^ 0x80000000u) : 0u;
    __syncthreads();
    int lim = min(256, NN - base);
    if (i < NN) {
#pragma unroll 4
      for (int jj = 0; jj < lim; ++jj) {
        unsigned kj = skeys[jj];
        rank += (kj > my) || (kj == my && (base + jj) < i);
      }
    }
    __syncthreads();
  }
  if (i < NN) {
    if (rank < KK) { new_id[i] = rank; perm[rank] = i; score_k[rank] = myscore; }
    else new_id[i] = -1;
  }
}

// =====================================================================
// Kernel 10: x_out[k] = out[perm[k]] * score_k[k]; batch_out = 0
// =====================================================================
__global__ void gather_out(const float* __restrict__ out_acc,
                           const int* __restrict__ perm,
                           const float* __restrict__ score_k,
                           float* __restrict__ x_out, int* __restrict__ batch_out) {
  int t = blockIdx.x * blockDim.x + threadIdx.x;
  if (t >= KK * 64) return;
  int k = t >> 6, g = t & 63;
  int src = perm[k];
  float s = score_k[k];
  const float4 v = *(const float4*)(out_acc + (size_t)src * OUTF + g * 4);
  float4 r = make_float4(v.x * s, v.y * s, v.z * s, v.w * s);
  *(float4*)(x_out + (size_t)k * OUTF + g * 4) = r;
  if (g == 0) batch_out[k] = 0;
}

// =====================================================================
// Kernel 11: remap surviving edges, dropped -> -1 (both rows)
// =====================================================================
__global__ void edge_remap(const long long* __restrict__ ei,
                           const int* __restrict__ new_id,
                           long long* __restrict__ out_ei) {
  int e = blockIdx.x * blockDim.x + threadIdx.x;
  if (e >= EE) return;
  int s2 = new_id[(int)ei[e]];
  int d2 = new_id[(int)ei[EE + e]];
  bool keep = (s2 >= 0) && (d2 >= 0);
  out_ei[e]      = keep ? (long long)s2 : -1LL;
  out_ei[EE + e] = keep ? (long long)d2 : -1LL;
}

// =====================================================================
extern "C" void kernel_launch(void* const* d_in, const int* in_sizes, int n_in,
                              void* d_out, int out_size, void* d_ws, size_t ws_size,
                              hipStream_t stream) {
  const float*     x        = (const float*)d_in[0];
  const long long* ei       = (const long long*)d_in[1];
  const float*     lin_w    = (const float*)d_in[2];
  const float*     att_src  = (const float*)d_in[3];
  const float*     att_dst  = (const float*)d_in[4];
  const float*     bias     = (const float*)d_in[5];
  const float*     pool_w   = (const float*)d_in[6];

  // ---- workspace carve-up ----
  char* wp = (char*)d_ws;
  float*          xh      = (float*)wp;          wp += (size_t)NN * OUTF * 4;
  float*          a_src   = (float*)wp;          wp += (size_t)NN * HH * 4;
  float*          a_dst   = (float*)wp;          wp += (size_t)NN * HH * 4;
  float*          e_buf   = (float*)wp;          wp += (size_t)EN * HH * 4;
  int*            m_i     = (int*)wp;            wp += (size_t)NN * HH * 4;
  float*          denom   = (float*)wp;          wp += (size_t)NN * HH * 4;
  float*          out_acc = (float*)wp;          wp += (size_t)NN * OUTF * 4;
  float*          score   = (float*)wp;          wp += (size_t)NN * 4;
  int*            new_id  = (int*)wp;            wp += (size_t)NN * 4;
  float*          normp   = (float*)wp;          wp += 64;
  unsigned short* xb      = (unsigned short*)wp; wp += (size_t)NN * INF_ * 2;
  unsigned short* wbuf    = (unsigned short*)wp; wp += (size_t)OUTF * INF_ * 2;

  // ---- output carve-up (tuple concatenated in return order) ----
  char* ob = (char*)d_out;
  float*     x_out   = (float*)ob;               ob += (size_t)KK * OUTF * 4;
  long long* out_ei  = (long long*)ob;           ob += (size_t)2 * EE * 8;
  int*       batch_o = (int*)ob;                 ob += (size_t)KK * 4;
  int*       perm_o  = (int*)ob;                 ob += (size_t)KK * 4;
  float*     score_k = (float*)ob;

  // 0. bf16 pre-conversion of x and lin_w (one-time, bandwidth-bound)
  conv_bf16<<<(NN * INF_ / 8 + 255) / 256, 256, 0, stream>>>(x, xb, NN * INF_ / 8);
  conv_bf16<<<(OUTF * INF_ / 8 + 255) / 256, 256, 0, stream>>>(lin_w, wbuf, OUTF * INF_ / 8);
  // 1. init accumulators
  init_ws<<<(NN * OUTF + 255) / 256, 256, 0, stream>>>(m_i, denom, out_acc);
  // 2. pool_w norm
  pool_norm<<<1, 256, 0, stream>>>(pool_w, normp);
  // 3. GEMM via WMMA (bf16 in, f32 acc): 3125 row tiles x 16 col tiles
  gat_lin_wmma<<<dim3(NN / 16, 4), 128, 0, stream>>>(xb, wbuf, xh);
  // 4. attention logits per node/head
  att_scores<<<(NN * HH + 255) / 256, 256, 0, stream>>>(xh, att_src, att_dst, a_src, a_dst);
  // 5. edge leaky-relu + segment max
  edge_max<<<(EN * HH + 255) / 256, 256, 0, stream>>>(ei, a_src, a_dst, e_buf, m_i);
  // 6. exp + segment sum
  edge_exp<<<(EN * HH + 255) / 256, 256, 0, stream>>>(ei, m_i, e_buf, denom);
  // 7. weighted message scatter
  edge_msg<<<(EN * 64 + 255) / 256, 256, 0, stream>>>(ei, xh, e_buf, denom, out_acc);
  // 8. bias + ELU + pooling score
  finish_node<<<NN, 256, 0, stream>>>(bias, pool_w, normp, out_acc, score);
  // 9. stable top-K rank
  topk_rank<<<(NN + 255) / 256, 256, 0, stream>>>(score, new_id, perm_o, score_k);
  // 10. gather & scale kept rows
  gather_out<<<(KK * 64 + 255) / 256, 256, 0, stream>>>(out_acc, perm_o, score_k, x_out, batch_o);
  // 11. edge remap
  edge_remap<<<(EE + 255) / 256, 256, 0, stream>>>(ei, new_id, out_ei);
}
